// TransformerAttentionBlock_13030930776306
// MI455X (gfx1250) — matrix operand order repaired and checked by reading, not compiled, not run on hardware
//
#include <hip/hip_runtime.h>

// ---------------------------------------------------------------------------
// CDNA5 / gfx1250 transformer attention block (LN -> QKV -> RoPE -> SWA -> proj)
// Matmuls via V_WMMA_F32_16X16X32_BF16 (bf16 in, f32 accumulate).
// GEMM: 128x128 block tile, LDS-staged double-buffered B; per-K-step issue
// order pinned with sched_group_barrier: VMEM(4) -> DSread(16) -> WMMA(8)
// -> DSwrite(2), so WMMAs overlap LDS/global latency instead of s_wait 0.
// ---------------------------------------------------------------------------

typedef __bf16 bf16_t;
typedef __attribute__((ext_vector_type(4)))  __bf16 v4bf;
typedef __attribute__((ext_vector_type(8)))  __bf16 v8bf;
typedef __attribute__((ext_vector_type(16))) __bf16 v16bf;
typedef __attribute__((ext_vector_type(8)))  float  v8f;

#define DIMC   1024
#define HEADS  16
#define DHEAD  64
#define TSEQ   2048
#define BATCH  2
#define ROWS   (BATCH * TSEQ)   // 4096 tokens
#define NQKV   (3 * DIMC)       // 3072
#define BROW   40               // padded LDS row stride (bf16) for B tiles

#if __has_builtin(__builtin_amdgcn_sched_group_barrier)
#define SGB(mask, size, id) __builtin_amdgcn_sched_group_barrier((mask), (size), (id))
#else
#define SGB(mask, size, id)
#endif

__device__ __forceinline__ v16bf cat16(v8bf lo, v8bf hi) {
  return __builtin_shufflevector(lo, hi, 0,1,2,3,4,5,6,7,8,9,10,11,12,13,14,15);
}
__device__ __forceinline__ v8f wmma_bf16(v16bf a, v16bf b, v8f c) {
  // D = A(16x32) * B(32x16) + C, f32 accumulate
  return __builtin_amdgcn_wmma_f32_16x16x32_bf16(false, a, false, b, (short)0, c,
                                                 false, false);
}

// ---------------------------------------------------------------------------
// K0: cast fp32 weights -> bf16 (w_qkv then w_out, contiguous in workspace)
// ---------------------------------------------------------------------------
__global__ void cast_weights_kernel(const float* __restrict__ wqkv,
                                    const float* __restrict__ wout,
                                    bf16_t* __restrict__ dst) {
  int i = blockIdx.x * blockDim.x + threadIdx.x;
  const int n1 = NQKV * DIMC;          // 3,145,728
  const int nt = n1 + DIMC * DIMC;     // 4,194,304
  if (i >= nt) return;
  float v = (i < n1) ? wqkv[i] : wout[i - n1];
  dst[i] = (bf16_t)v;
}

// ---------------------------------------------------------------------------
// K1: LayerNorm (one 1024-wide row per 256-thread block) -> bf16 xn
// ---------------------------------------------------------------------------
__global__ void layernorm_kernel(const float* __restrict__ x,
                                 const float* __restrict__ w,
                                 const float* __restrict__ b,
                                 bf16_t* __restrict__ xn) {
  const int row  = blockIdx.x;
  const int tid  = threadIdx.x;           // 0..255, 4 elems each
  const int lane = tid & 31;
  const int wid  = tid >> 5;
  const float* xr = x + (size_t)row * DIMC;
  float4 v = ((const float4*)xr)[tid];

  __shared__ float red[8];
  float s = v.x + v.y + v.z + v.w;
#pragma unroll
  for (int off = 16; off; off >>= 1) s += __shfl_xor(s, off, 32);
  if (!lane) red[wid] = s;
  __syncthreads();
  float tot = 0.f;
#pragma unroll
  for (int i = 0; i < 8; i++) tot += red[i];
  const float mean = tot * (1.f / DIMC);

  float dx = v.x - mean, dy = v.y - mean, dz = v.z - mean, dw = v.w - mean;
  float s2 = dx*dx + dy*dy + dz*dz + dw*dw;
#pragma unroll
  for (int off = 16; off; off >>= 1) s2 += __shfl_xor(s2, off, 32);
  __syncthreads();
  if (!lane) red[wid] = s2;
  __syncthreads();
  float tot2 = 0.f;
#pragma unroll
  for (int i = 0; i < 8; i++) tot2 += red[i];
  const float inv = rsqrtf(tot2 * (1.f / DIMC) + 1e-5f);

  float4 wv = ((const float4*)w)[tid];
  float4 bv = ((const float4*)b)[tid];
  v4bf o;
  o[0] = (bf16_t)(dx * inv * wv.x + bv.x);
  o[1] = (bf16_t)(dy * inv * wv.y + bv.y);
  o[2] = (bf16_t)(dz * inv * wv.z + bv.z);
  o[3] = (bf16_t)(dw * inv * wv.w + bv.w);
  *(v4bf*)(xn + (size_t)row * DIMC + tid * 4) = o;
}

// ---------------------------------------------------------------------------
// K2/K5: bf16 WMMA GEMM  Y[M,N] = A[M,K] * W[N,K]^T   (+ optional residual)
// 256 threads = 8 waves; block tile 128(M) x 128(N); wave tile 16 x 128.
// ---------------------------------------------------------------------------
__global__ void gemm_bf16_kernel(const bf16_t* __restrict__ A,
                                 const bf16_t* __restrict__ W,
                                 bf16_t* __restrict__ Obf,
                                 float* __restrict__ Of32,
                                 const float* __restrict__ resid,
                                 int M, int N, int K) {
  __shared__ __align__(16) bf16_t ldsB[2][128 * BROW];   // 2 x 10 KB
  const int tid  = threadIdx.x;
  const int lane = tid & 31;
  const int wave = tid >> 5;               // 0..7 -> M sub-tile
  const int mrow = lane & 15;
  const int hi   = lane >> 4;
  const int nb   = N >> 7;                 // N / 128
  const int bm   = blockIdx.x / nb;
  const int bn   = blockIdx.x % nb;

  // B staging: each thread owns two 16B chunks of the 128x32 B tile.
  const int brow = tid >> 1;               // 0..127 (row within N tile)
  const int bko  = (tid & 1) * 16;         // 0 or 16 (bf16 elems along K)
  const bf16_t* wChunk = W + (size_t)(bn * 128 + brow) * K + bko;
  bf16_t* ldsDst0 = &ldsB[0][brow * BROW + bko];
  bf16_t* ldsDst1 = &ldsB[1][brow * BROW + bko];

  const bf16_t* aRow = A + (size_t)(bm * 128 + wave * 16 + mrow) * K;

  // ---- prologue: stage k-step 0 into buffer 0 ----
  {
    v8bf b0 = *(const v8bf*)(wChunk);
    v8bf b1 = *(const v8bf*)(wChunk + 8);
    *(v8bf*)(ldsDst0)     = b0;
    *(v8bf*)(ldsDst0 + 8) = b1;
  }
  v8bf a0n = *(const v8bf*)(aRow + 16 * hi);
  v8bf a1n = *(const v8bf*)(aRow + 16 * hi + 8);
  __syncthreads();

  v8f acc[8];
#pragma unroll
  for (int i = 0; i < 8; i++) acc[i] = v8f{};

  const int NSTEP = K >> 5;
  for (int ks = 0; ks < NSTEP; ks++) {
    const int  cur  = ks & 1;
    const bool more = (ks + 1) < NSTEP;
    const int  kn   = (more ? ks + 1 : ks) << 5;

    // current A fragment (loaded last iteration)
    v16bf afrag = cat16(a0n, a1n);

    // issue next k-step's global loads (overlap with WMMA below)
    v8bf nb0 = *(const v8bf*)(wChunk + kn);
    v8bf nb1 = *(const v8bf*)(wChunk + kn + 8);
    a0n = *(const v8bf*)(aRow + kn + 16 * hi);
    a1n = *(const v8bf*)(aRow + kn + 16 * hi + 8);

    // batch-issue all 16 ds_load_b128 for this K-step, then 8 WMMAs
    const bf16_t* Bl = ldsB[cur] + (size_t)mrow * BROW + 16 * hi;
    v8bf f0[8], f1[8];
#pragma unroll
    for (int j = 0; j < 8; j++) {
      const bf16_t* p = Bl + j * (16 * BROW);
      f0[j] = *(const v8bf*)(p);
      f1[j] = *(const v8bf*)(p + 8);
    }
#pragma unroll
    for (int j = 0; j < 8; j++)
      acc[j] = wmma_bf16(afrag, cat16(f0[j], f1[j]), acc[j]);

    // stage next B tile into the other buffer
    if (more) {
      bf16_t* dst = cur ? ldsDst0 : ldsDst1;
      *(v8bf*)(dst)     = nb0;
      *(v8bf*)(dst + 8) = nb1;
    }

    // Pin the machine-scheduler issue order for this K-step:
    // 4 global reads -> 16 DS reads -> 8 WMMA -> 2 DS writes.
    SGB(0x020, 4, 0);    // VMEM reads (next B chunks + next A fragment)
    SGB(0x100, 16, 0);   // DS reads (all 8 B fragments)
    SGB(0x008, 8, 0);    // WMMA, back-to-back
    SGB(0x200, 2, 0);    // DS writes (stage next buffer)

    __syncthreads();
  }

  // Epilogue. C layout: lane, vgpr r -> (M = r + 8*hi, N = lane&15)
#pragma unroll
  for (int j = 0; j < 8; j++) {
#pragma unroll
    for (int r = 0; r < 8; r++) {
      const size_t row = (size_t)(bm * 128 + wave * 16 + r + 8 * hi);
      const size_t col = (size_t)(bn * 128 + j * 16 + mrow);
      if (Of32) {
        Of32[row * N + col] = acc[j][r] + resid[row * N + col];
      } else {
        Obf[row * N + col] = (bf16_t)acc[j][r];
      }
    }
  }
}

// ---------------------------------------------------------------------------
// K3: RoPE + head split. qkv bf16 [ROWS, 3072] ->
//     Q,K bf16 [B,H,T,64] row-major ; V bf16 transposed [B,H,64,T]
// One wave per (b,t,h); lane = rotation pair index d (0..31).
// ---------------------------------------------------------------------------
__global__ void rope_kernel(const bf16_t* __restrict__ qkv,
                            bf16_t* __restrict__ Qo,
                            bf16_t* __restrict__ Ko,
                            bf16_t* __restrict__ Vt) {
  const int lane = threadIdx.x & 31;
  const int wv   = threadIdx.x >> 5;
  const int idx  = blockIdx.x * 4 + wv;       // (b,t,h): 2*2048*16 = 65536
  const int b = idx >> 15;
  const int t = (idx >> 4) & (TSEQ - 1);
  const int h = idx & (HEADS - 1);

  const size_t base = (size_t)(b * TSEQ + t) * NQKV + h * DHEAD;
  float qlo = (float)qkv[base + lane];
  float qhi = (float)qkv[base + 32 + lane];
  float klo = (float)qkv[base + DIMC + lane];
  float khi = (float)qkv[base + DIMC + 32 + lane];
  float vlo = (float)qkv[base + 2 * DIMC + lane];
  float vhi = (float)qkv[base + 2 * DIMC + 32 + lane];

  // inv_freq[d] = 10000^(-d/32) = exp(-d * ln(10000)/32)
  float freq = __expf(-(float)lane * 0.28782313662425574f);
  float th = (float)t * freq;
  float c = cosf(th), s = sinf(th);

  const size_t obase = ((size_t)(b * HEADS + h) * TSEQ + t) * DHEAD;
  Qo[obase + lane]      = (bf16_t)(qlo * c - qhi * s);
  Qo[obase + 32 + lane] = (bf16_t)(qhi * c + qlo * s);
  Ko[obase + lane]      = (bf16_t)(klo * c - khi * s);
  Ko[obase + 32 + lane] = (bf16_t)(khi * c + klo * s);

  const size_t vbase = (size_t)(b * HEADS + h) * DHEAD * TSEQ;
  Vt[vbase + (size_t)lane * TSEQ + t]        = (bf16_t)vlo;
  Vt[vbase + (size_t)(lane + 32) * TSEQ + t] = (bf16_t)vhi;
}

// ---------------------------------------------------------------------------
// K4: sliding-window attention. One wave per (b,h,16-query tile).
// Keys needed: [qbase-32, qbase+15] -> 3 key tiles of 16.
// S = Q K^T (6 WMMA), shfl softmax, P restaged via LDS, O = P V (8 WMMA).
// ---------------------------------------------------------------------------
__global__ void attn_kernel(const bf16_t* __restrict__ Q,
                            const bf16_t* __restrict__ Km,
                            const bf16_t* __restrict__ Vt,
                            bf16_t* __restrict__ Out) {
  __shared__ __align__(16) bf16_t ldsP[4][16 * 64];   // per-wave P tile, padded
  const int lane = threadIdx.x & 31;
  const int wave = threadIdx.x >> 5;
  const int tile = blockIdx.x * 4 + wave;      // 0..4095
  const int bh    = tile >> 7;                 // T/16 = 128 q-tiles per head
  const int qbase = (tile & 127) * 16;
  const int mrow  = lane & 15;
  const int hi    = lane >> 4;

  const bf16_t* Qp = Q  + (size_t)bh * TSEQ * DHEAD;
  const bf16_t* Kp = Km + (size_t)bh * TSEQ * DHEAD;
  const bf16_t* Vp = Vt + (size_t)bh * DHEAD * TSEQ;

  // --- Q fragments (headdim 64 = two K=32 steps) ---
  const bf16_t* qr = Qp + (size_t)(qbase + mrow) * DHEAD;
  v16bf aQ[2];
#pragma unroll
  for (int s = 0; s < 2; s++) {
    v8bf q0 = *(const v8bf*)(qr + s * 32 + 16 * hi);
    v8bf q1 = *(const v8bf*)(qr + s * 32 + 16 * hi + 8);
    aQ[s] = cat16(q0, q1);
  }

  // --- scores: 3 key tiles ---
  const int kbase = qbase - 32;
  v8f S[3] = {v8f{}, v8f{}, v8f{}};
#pragma unroll
  for (int kt = 0; kt < 3; kt++) {
    int key  = kbase + kt * 16 + mrow;
    int keyc = key < 0 ? 0 : key;          // clamped load; mask kills garbage
#pragma unroll
    for (int s = 0; s < 2; s++) {
      v8bf b0 = *(const v8bf*)(Kp + (size_t)keyc * DHEAD + s * 32 + 16 * hi);
      v8bf b1 = *(const v8bf*)(Kp + (size_t)keyc * DHEAD + s * 32 + 16 * hi + 8);
      S[kt] = wmma_bf16(aQ[s], cat16(b0, b1), S[kt]);
    }
  }

  // --- mask + softmax (rows span the 16 lanes of each half-wave) ---
  const float scale = 0.125f;  // 1/sqrt(64)
  float pv[3][8], rinv[8];
#pragma unroll
  for (int r = 0; r < 8; r++) {
    const int mq = qbase + r + 8 * hi;
    float mx = -1e30f;
#pragma unroll
    for (int kt = 0; kt < 3; kt++) {
      int nk = kbase + kt * 16 + mrow;
      bool valid = (nk >= 0) && (mq >= nk) && (mq - nk < 32);
      float sv = valid ? S[kt][r] * scale : -1e30f;
      pv[kt][r] = sv;
      mx = fmaxf(mx, sv);
    }
#pragma unroll
    for (int off = 1; off < 16; off <<= 1) mx = fmaxf(mx, __shfl_xor(mx, off, 32));
    float sum = 0.f;
#pragma unroll
    for (int kt = 0; kt < 3; kt++) {
      float e = __expf(pv[kt][r] - mx);
      pv[kt][r] = e;
      sum += e;
    }
#pragma unroll
    for (int off = 1; off < 16; off <<= 1) sum += __shfl_xor(sum, off, 32);
    rinv[r] = 1.f / sum;
  }

  // --- restage P into LDS as [16 q][64 k] (cols 48..63 zero) ---
  bf16_t* Pl = ldsP[wave];
#pragma unroll
  for (int r = 0; r < 8; r++) {
    const int row = r + 8 * hi;
#pragma unroll
    for (int kt = 0; kt < 3; kt++)
      Pl[row * 64 + kt * 16 + mrow] = (bf16_t)(pv[kt][r] * rinv[r]);
    Pl[row * 64 + 48 + mrow] = (bf16_t)0.f;
  }
  __syncthreads();

  // --- O = P V : 2 key groups x 4 dim tiles ---
  v8f O[4] = {v8f{}, v8f{}, v8f{}, v8f{}};
#pragma unroll
  for (int g = 0; g < 2; g++) {
    v8bf p0 = *(const v8bf*)(Pl + mrow * 64 + g * 32 + 16 * hi);
    v8bf p1 = *(const v8bf*)(Pl + mrow * 64 + g * 32 + 16 * hi + 8);
    v16bf aP = cat16(p0, p1);
#pragma unroll
    for (int j = 0; j < 4; j++) {
      const int dim = j * 16 + mrow;
      int k0 = kbase + g * 32 + 16 * hi;
      int k0a = k0 < 0 ? 0 : k0;
      int k0b = (k0 + 8) < 0 ? 0 : (k0 + 8);
      v8bf b0 = *(const v8bf*)(Vp + (size_t)dim * TSEQ + k0a);
      v8bf b1 = *(const v8bf*)(Vp + (size_t)dim * TSEQ + k0b);
      O[j] = wmma_bf16(aP, cat16(b0, b1), O[j]);
    }
  }

  // --- store to [B,T,H*64] bf16 for the output projection ---
  const int b = bh >> 4, h = bh & 15;
#pragma unroll
  for (int j = 0; j < 4; j++) {
#pragma unroll
    for (int r = 0; r < 8; r++) {
      const size_t trow = (size_t)(b * TSEQ + qbase + r + 8 * hi);
      const size_t col  = (size_t)(h * DHEAD + j * 16 + mrow);
      Out[trow * DIMC + col] = (bf16_t)O[j][r];
    }
  }
}

// ---------------------------------------------------------------------------
// Host launcher
// ---------------------------------------------------------------------------
extern "C" void kernel_launch(void* const* d_in, const int* in_sizes, int n_in,
                              void* d_out, int out_size, void* d_ws, size_t ws_size,
                              hipStream_t stream) {
  const float* x     = (const float*)d_in[0];
  const float* ln_w  = (const float*)d_in[1];
  const float* ln_b  = (const float*)d_in[2];
  const float* w_qkv = (const float*)d_in[3];
  const float* w_out = (const float*)d_in[4];
  float* out = (float*)d_out;

  char* ws = (char*)d_ws;
  // workspace layout (bytes)
  bf16_t* ws_wqkv = (bf16_t*)(ws + 0);                       // 3072*1024*2 = 6291456
  bf16_t* ws_wout = (bf16_t*)(ws + 6291456);                 // 1024*1024*2 = 2097152
  bf16_t* ws_xn   = (bf16_t*)(ws + 8388608);                 // 4096*1024*2 = 8388608
  bf16_t* ws_qkv  = (bf16_t*)(ws + 16777216);                // 4096*3072*2 = 25165824
  bf16_t* ws_q    = (bf16_t*)(ws + 41943040);                // 8388608
  bf16_t* ws_k    = (bf16_t*)(ws + 50331648);                // 8388608
  bf16_t* ws_vt   = (bf16_t*)(ws + 58720256);                // 8388608
  bf16_t* ws_attn = (bf16_t*)(ws + 67108864);                // 8388608
  // total: 75497472 bytes

  // K0: weights -> bf16
  cast_weights_kernel<<<16384, 256, 0, stream>>>(w_qkv, w_out, ws_wqkv);
  // K1: layernorm -> bf16 xn
  layernorm_kernel<<<ROWS, 256, 0, stream>>>(x, ln_w, ln_b, ws_xn);
  // K2: qkv = xn @ Wqkv^T   (block tiles: (4096/128)*(3072/128) = 768)
  gemm_bf16_kernel<<<768, 256, 0, stream>>>(ws_xn, ws_wqkv, ws_qkv, nullptr,
                                            nullptr, ROWS, NQKV, DIMC);
  // K3: RoPE + reshape (+ V transpose)
  rope_kernel<<<16384, 128, 0, stream>>>(ws_qkv, ws_q, ws_k, ws_vt);
  // K4: sliding-window attention (4096 q-tiles, 4 waves/block)
  attn_kernel<<<1024, 128, 0, stream>>>(ws_q, ws_k, ws_vt, ws_attn);
  // K5: out = attn @ Wout^T + x   (block tiles: (4096/128)*(1024/128) = 256)
  gemm_bf16_kernel<<<256, 256, 0, stream>>>(ws_attn, ws_wout, nullptr, out,
                                            x, ROWS, DIMC, DIMC);
}
